// SimplifiedSSMBlock_41652592836926
// MI455X (gfx1250) — matrix operand order repaired and checked by reading, not compiled, not run on hardware
//
#include <hip/hip_runtime.h>
#include <hip/hip_bf16.h>
#include <math.h>

typedef __attribute__((ext_vector_type(16))) _Float16 v16h;
typedef __attribute__((ext_vector_type(8)))  _Float16 v8h;
typedef __attribute__((ext_vector_type(8)))  float    v8f;

// ---------------- epilogue ids ----------------
#define EPI_F32      0   // store f32
#define EPI_BOTH     1   // store f32 + f16
#define EPI_SILU     2   // store silu(x) f32
#define EPI_SOFTPLUS 3   // store clip(softplus(x + bias[n]), 1e-3, 1) f32
#define EPI_RESID    4   // store resid[idx] + x f32

// ---------------- LayerNorm -> f16 ----------------
__global__ void ln_kernel(const float* __restrict__ x,
                          const float* __restrict__ w,
                          const float* __restrict__ b,
                          _Float16* __restrict__ xnh, int DM) {
  __shared__ float red[256];
  const size_t row = blockIdx.x;
  const float* xr = x + row * (size_t)DM;
  float s = 0.f;
  for (int i = threadIdx.x; i < DM; i += blockDim.x) s += xr[i];
  red[threadIdx.x] = s; __syncthreads();
  for (int off = 128; off > 0; off >>= 1) {
    if ((int)threadIdx.x < off) red[threadIdx.x] += red[threadIdx.x + off];
    __syncthreads();
  }
  const float mu = red[0] / (float)DM;
  __syncthreads();
  float v = 0.f;
  for (int i = threadIdx.x; i < DM; i += blockDim.x) {
    float d = xr[i] - mu; v += d * d;
  }
  red[threadIdx.x] = v; __syncthreads();
  for (int off = 128; off > 0; off >>= 1) {
    if ((int)threadIdx.x < off) red[threadIdx.x] += red[threadIdx.x + off];
    __syncthreads();
  }
  const float inv = rsqrtf(red[0] / (float)DM + 1e-5f);
  for (int i = threadIdx.x; i < DM; i += blockDim.x)
    xnh[row * (size_t)DM + i] = (_Float16)((xr[i] - mu) * inv * w[i] + b[i]);
}

// ------- transpose + convert: in f32 [K x N] -> out f16 [N x K] -------
// out may point inside a larger matrix (row stride = outStride).
__global__ void transpose_f16_kernel(const float* __restrict__ in,
                                     _Float16* __restrict__ out,
                                     int K, int N, int outStride) {
  __shared__ float tile[16][17];  // +1 pad: no bank conflicts
  const int kb = blockIdx.x * 16, nb = blockIdx.y * 16;
  const int tx = threadIdx.x, ty = threadIdx.y;
  tile[ty][tx] = in[(size_t)(kb + ty) * N + (nb + tx)];
  __syncthreads();
  out[(size_t)(nb + ty) * outStride + (kb + tx)] = (_Float16)tile[tx][ty];
}

// ---------------- WMMA fragments as contiguous 128-bit loads ----------------
__device__ __forceinline__ v16h cat8(v8h lo, v8h hi) {
  return __builtin_shufflevector(lo, hi, 0, 1, 2, 3, 4, 5, 6, 7,
                                 8, 9, 10, 11, 12, 13, 14, 15);
}
// A fragment (16x32 f16, MxK, row-major): lane lm -> row, half -> +8*half:
//   f[0..7]  = A[row][k0+8h .. +7], f[8..15] = A[row][k0+8h+16 .. +23]
__device__ __forceinline__ v16h load_a_frag(const _Float16* __restrict__ arow,
                                            int k0, int half) {
  const _Float16* p = arow + k0 + 8 * half;
  return cat8(*(const v8h*)p, *(const v8h*)(p + 16));
}
// B fragment (32x16 f16, KxN) from transposed weights Bt[N x K]: lane lm ->
// column n; f[0..7] = Bt[n][k0+8h .. +7], f[8..15] = Bt[n][k0+8h+16 .. +23].
__device__ __forceinline__ v16h load_b_frag(const _Float16* __restrict__ brow,
                                            int k0, int half) {
  const _Float16* p = brow + k0 + 8 * half;
  return cat8(*(const v8h*)p, *(const v8h*)(p + 16));
}

// ------- WMMA GEMM: C(MxN) = A(MxK) @ B(KxN) with B given transposed -------
// block = 128 threads (4 waves); block tile = 128 rows x (16*NTILES) cols;
// wave w -> rows [blk*128 + 32w, +32) (two 16-row tiles), NTILES 16-col tiles.
template <int EPI, int NTILES>
__global__ __launch_bounds__(128)
void wmma_gemm_kernel(const _Float16* __restrict__ A,
                      const _Float16* __restrict__ Bt,  // N x K row-major
                      float* __restrict__ Cf,
                      _Float16* __restrict__ Ch,
                      const float* __restrict__ bias,
                      const float* __restrict__ resid,
                      int M, int N, int K) {
  const int wave  = threadIdx.x >> 5;
  const int lane  = threadIdx.x & 31;
  const int lm    = lane & 15;
  const int half  = lane >> 4;
  const int mbase = blockIdx.x * 128 + wave * 32;
  const int nbase = blockIdx.y * (16 * NTILES);

  v8f acc[2][NTILES];
#pragma unroll
  for (int mt = 0; mt < 2; ++mt)
#pragma unroll
    for (int nt = 0; nt < NTILES; ++nt) acc[mt][nt] = v8f{};

  const _Float16* arow0 = A + (size_t)(mbase + lm) * K;
  const _Float16* arow1 = A + (size_t)(mbase + 16 + lm) * K;
  const _Float16* brow[NTILES];
#pragma unroll
  for (int nt = 0; nt < NTILES; ++nt)
    brow[nt] = Bt + (size_t)(nbase + nt * 16 + lm) * K;

  for (int k0 = 0; k0 < K; k0 += 32) {
    if (k0 + 32 < K) {
      __builtin_prefetch(arow0 + k0 + 32, 0, 3);
      __builtin_prefetch(arow1 + k0 + 32, 0, 3);
    }
    const v16h a0 = load_a_frag(arow0, k0, half);
    const v16h a1 = load_a_frag(arow1, k0, half);
#pragma unroll
    for (int nt = 0; nt < NTILES; ++nt) {
      const v16h b = load_b_frag(brow[nt], k0, half);
      acc[0][nt] = __builtin_amdgcn_wmma_f32_16x16x32_f16(
          false, a0, false, b, (short)0, acc[0][nt], false, false);
      acc[1][nt] = __builtin_amdgcn_wmma_f32_16x16x32_f16(
          false, a1, false, b, (short)0, acc[1][nt], false, false);
    }
  }

  // C/D f32 layout: lane n = lm; rows mtile + 8*half + i in VGPR i.
#pragma unroll
  for (int mt = 0; mt < 2; ++mt) {
#pragma unroll
    for (int nt = 0; nt < NTILES; ++nt) {
      const int n = nbase + nt * 16 + lm;
#pragma unroll
      for (int i = 0; i < 8; ++i) {
        const int m = mbase + mt * 16 + half * 8 + i;
        const size_t idx = (size_t)m * N + n;
        float v = acc[mt][nt][i];
        if (EPI == EPI_SILU) {
          Cf[idx] = v / (1.f + __expf(-v));
        } else if (EPI == EPI_SOFTPLUS) {
          v += bias[n];
          float sp = (v > 20.f) ? v : log1pf(__expf(v));
          Cf[idx] = fminf(fmaxf(sp, 1e-3f), 1.0f);
        } else if (EPI == EPI_RESID) {
          Cf[idx] = resid[idx] + v;
        } else {
          Cf[idx] = v;
          if (EPI == EPI_BOTH) Ch[idx] = (_Float16)v;
        }
      }
    }
  }
}

// ---------------- sequential selective scan, fused gate ----------------
// BC: [rows x 32] fused (B | C) projections; each lane owns channel d.
__global__ void scan_kernel(const float* __restrict__ dt,
                            const float* __restrict__ z,
                            const float* __restrict__ BC,
                            const float* __restrict__ gate,
                            const float* __restrict__ A_log,
                            _Float16* __restrict__ yg,
                            int Tlen, int DI) {
  const int b = blockIdx.y;
  const int d = blockIdx.x * blockDim.x + threadIdx.x;
  __shared__ float sBC[32];  // [0..15]=B_t, [16..31]=C_t

  float h[16], Ar[16];
#pragma unroll
  for (int s = 0; s < 16; ++s) {
    h[s] = 0.f;
    float a = __expf(A_log[(size_t)d * 16 + s]);
    Ar[s] = -fminf(fmaxf(a, 1e-6f), 1.0f);
  }

  const size_t rowbase = (size_t)b * Tlen;
  for (int t = 0; t < Tlen; ++t) {
    const size_t row = rowbase + t;
    if (threadIdx.x < 32) sBC[threadIdx.x] = BC[row * 32 + threadIdx.x];
    __syncthreads();

    const size_t idx = row * (size_t)DI + d;
    const float dtv = dt[idx];
    const float zv  = z[idx];
    float y = 0.f;
#pragma unroll
    for (int s = 0; s < 16; ++s) {
      const float da = __expf(fmaxf(dtv * Ar[s], -20.f));  // dt*A <= 0 always
      h[s] = da * h[s] + (dtv * sBC[s]) * zv;
      y += h[s] * sBC[16 + s];
    }
    yg[idx] = (_Float16)(y * gate[idx]);
    __syncthreads();
  }
}

// ---------------- host launcher ----------------
extern "C" void kernel_launch(void* const* d_in, const int* in_sizes, int n_in,
                              void* d_out, int out_size, void* d_ws, size_t ws_size,
                              hipStream_t stream) {
  const float* x      = (const float*)d_in[0];
  const float* norm_w = (const float*)d_in[1];
  const float* norm_b = (const float*)d_in[2];
  const float* W_in   = (const float*)d_in[3];
  const float* W_gate = (const float*)d_in[4];
  const float* W_dt   = (const float*)d_in[5];
  const float* b_dt   = (const float*)d_in[6];
  const float* W_B    = (const float*)d_in[7];
  const float* W_C    = (const float*)d_in[8];
  const float* A_log  = (const float*)d_in[9];
  const float* W_out  = (const float*)d_in[10];
  float* out = (float*)d_out;

  const int DM   = in_sizes[1];          // 1024
  const int DI   = in_sizes[6];          // 2048
  const int rows = in_sizes[0] / DM;     // B*T = 8192
  const int DS   = in_sizes[7] / DI;     // 16
  const int BATCH = 2;
  const int T = rows / BATCH;
  (void)DS; (void)n_in; (void)out_size; (void)ws_size;

  // bump allocator over d_ws
  char* base = (char*)d_ws;
  size_t off = 0;
  auto alloc = [&](size_t bytes) -> void* {
    void* r = base + off;
    off = (off + bytes + 255) & ~(size_t)255;
    return r;
  };
  _Float16* xnh    = (_Float16*)alloc((size_t)rows * DM * 2);
  _Float16* WinT   = (_Float16*)alloc((size_t)DM * DI * 2);   // DI x DM
  _Float16* WgateT = (_Float16*)alloc((size_t)DM * DI * 2);   // DI x DM
  _Float16* WdtT   = (_Float16*)alloc((size_t)DI * DI * 2);   // DI x DI
  _Float16* WBCT   = (_Float16*)alloc((size_t)DI * 32 * 2);   // 32 x DI fused
  _Float16* WoutT  = (_Float16*)alloc((size_t)DI * DM * 2);   // DM x DI
  float*    z_f    = (float*)   alloc((size_t)rows * DI * 4);
  _Float16* z_h    = (_Float16*)alloc((size_t)rows * DI * 2);
  float*    gate_f = (float*)   alloc((size_t)rows * DI * 4);
  float*    dt_f   = (float*)   alloc((size_t)rows * DI * 4);
  float*    BC_f   = (float*)   alloc((size_t)rows * 32 * 4);
  _Float16* yg_h   = (_Float16*)alloc((size_t)rows * DI * 2);

  // 1) LayerNorm -> f16 activations
  ln_kernel<<<rows, 256, 0, stream>>>(x, norm_w, norm_b, xnh, DM);

  // 2) weight transpose+convert: W[K x N] f32 -> Wt[N x K] f16
  const dim3 tblk(16, 16);
  auto tr = [&](const float* src, _Float16* dst, int K, int N, int ldo) {
    transpose_f16_kernel<<<dim3(K / 16, N / 16), tblk, 0, stream>>>(src, dst, K, N, ldo);
  };
  tr(W_in,   WinT,   DM, DI, DM);
  tr(W_gate, WgateT, DM, DI, DM);
  tr(W_dt,   WdtT,   DI, DI, DI);
  tr(W_B,    WBCT,            DI, 16, DI);   // rows 0..15  of fused block
  tr(W_C,    WBCT + 16 * DI,  DI, 16, DI);   // rows 16..31 of fused block
  tr(W_out,  WoutT,  DI, DM, DI);

  const dim3 blk(128);

  // 3) z = xn @ W_in  (f32 for scan, f16 for downstream GEMMs)
  wmma_gemm_kernel<EPI_BOTH, 4><<<dim3(rows / 128, DI / 64), blk, 0, stream>>>(
      xnh, WinT, z_f, z_h, nullptr, nullptr, rows, DI, DM);
  // 4) gate = silu(xn @ W_gate)
  wmma_gemm_kernel<EPI_SILU, 4><<<dim3(rows / 128, DI / 64), blk, 0, stream>>>(
      xnh, WgateT, gate_f, nullptr, nullptr, nullptr, rows, DI, DM);
  // 5) dt = clip(softplus(z @ W_dt + b_dt), 1e-3, 1)
  wmma_gemm_kernel<EPI_SOFTPLUS, 4><<<dim3(rows / 128, DI / 64), blk, 0, stream>>>(
      z_h, WdtT, dt_f, nullptr, b_dt, nullptr, rows, DI, DI);
  // 6) [Bp | Cp] = z @ [W_B | W_C]  (fused, N = 32)
  wmma_gemm_kernel<EPI_F32, 2><<<dim3(rows / 128, 1), blk, 0, stream>>>(
      z_h, WBCT, BC_f, nullptr, nullptr, nullptr, rows, 32, DI);

  // 7) sequential scan, fused * gate, emit f16 for the output GEMM
  scan_kernel<<<dim3(DI / 256, BATCH), 256, 0, stream>>>(
      dt_f, z_f, BC_f, gate_f, A_log, yg_h, T, DI);

  // 8) out = residual + (y*gate) @ W_out
  wmma_gemm_kernel<EPI_RESID, 4><<<dim3(rows / 128, DM / 64), blk, 0, stream>>>(
      yg_h, WoutT, out, nullptr, nullptr, x, rows, DM, DI);
}
